// GCNTransformer_36756330119355
// MI455X (gfx1250) — compile-verified, matrix operand order checked by reading, not run on hardware
//
#include <hip/hip_runtime.h>

#define Bb   8
#define Ss   1024
#define Dd   768
#define Hh   8
#define DH   96
#define KNN  5
#define LL   3

typedef unsigned short ushort_t;
typedef unsigned int   uint_t;
typedef __attribute__((ext_vector_type(16))) __bf16 bf16x16;
typedef __attribute__((ext_vector_type(8)))  __bf16 bf16x8;
typedef __attribute__((ext_vector_type(8)))  float  floatx8;

__device__ inline ushort_t f2bf(float f) {
    uint_t u = __float_as_uint(f);
    u += 0x7FFFu + ((u >> 16) & 1u);          // round-to-nearest-even
    return (ushort_t)(u >> 16);
}

__device__ inline floatx8 wmma_bf16(bf16x16 a, bf16x16 b, floatx8 c) {
    return __builtin_amdgcn_wmma_f32_16x16x32_bf16(
        false, a, false, b, (short)0, c, false, false);
}

// Load one 16x32 A/B^T fragment from LDS (row-major, LDW ushorts per row).
template<int LDW>
__device__ inline bf16x16 ldsFrag(const ushort_t* T, int rowbase, int kchunk, int lane) {
    int r  = rowbase + (lane & 15);
    int kb = kchunk + ((lane >> 4) << 3);
    const __bf16* p = (const __bf16*)(T + r * LDW + kb);
    bf16x8 lo = *(const bf16x8*)(p);
    bf16x8 hi = *(const bf16x8*)(p + 16);
    bf16x16 f;
#pragma unroll
    for (int i = 0; i < 8; ++i) { f[i] = lo[i]; f[i + 8] = hi[i]; }
    return f;
}

// 16-lane (DPP row) reductions via row_ror:1/2/4/8 — VALU only, no LDS traffic.
template<int CTRL>
__device__ inline float dppRor(float v) {
    return __int_as_float(__builtin_amdgcn_update_dpp(
        0, __float_as_int(v), CTRL, 0xf, 0xf, true));
}
__device__ inline float halfMax(float v) {
    v = fmaxf(v, dppRor<0x121>(v));   // ROW_ROR:1
    v = fmaxf(v, dppRor<0x122>(v));   // ROW_ROR:2
    v = fmaxf(v, dppRor<0x124>(v));   // ROW_ROR:4
    v = fmaxf(v, dppRor<0x128>(v));   // ROW_ROR:8
    return v;
}
__device__ inline float halfSum(float v) {
    v += dppRor<0x121>(v);
    v += dppRor<0x122>(v);
    v += dppRor<0x124>(v);
    v += dppRor<0x128>(v);
    return v;
}

// ---------------------------------------------------------------------------
// bf16-operand GEMM: C[M,N] = A[M,K]*B (+bias) (+Cf if accum)
//   BMODE 0 (NT): B given as B^T row-major [N,K]  (x @ W^T, x @ x^T)
//   BMODE 1 (NN): B given row-major [K,N]         (An @ G)
// Outputs: Cf (f32, optional) and/or Cb (bf16, optional).
// 128 threads = 4 waves (2x2); block tile 64x128; each wave 32x64 -> 8 WMMA/kstep.
// ---------------------------------------------------------------------------
template<int BMODE>
__global__ __launch_bounds__(128) void gemm_bf16_kernel(
    const ushort_t* __restrict__ A, const ushort_t* __restrict__ Bm,
    float* __restrict__ Cf, ushort_t* __restrict__ Cb,
    const float* __restrict__ bias,
    int lda, int ldb, int ldc, int K,
    long sA, long sB, long sC, int accum)
{
    __shared__ ushort_t As[64 * 40];
    __shared__ ushort_t Bs[128 * 40];

    A  += (size_t)blockIdx.z * sA;
    Bm += (size_t)blockIdx.z * sB;
    const size_t cbase = (size_t)blockIdx.z * sC;

    const int tileN = blockIdx.x * 128, tileM = blockIdx.y * 64;
    const int tid = threadIdx.x, lane = tid & 31, wave = tid >> 5;
    const int wm = (wave >> 1) * 32, wn = (wave & 1) * 64;

    floatx8 z = {0.f,0.f,0.f,0.f,0.f,0.f,0.f,0.f};
    floatx8 c[2][4];
#pragma unroll
    for (int mi = 0; mi < 2; ++mi)
#pragma unroll
        for (int ni = 0; ni < 4; ++ni) c[mi][ni] = z;

    for (int k0 = 0; k0 < K; k0 += 32) {
#pragma unroll
        for (int i = 0; i < 2; ++i) {               // A tile 64x32 (8 bf16 chunks)
            int e = i * 128 + tid, r = e >> 2, cc = e & 3;
            *(uint4*)&As[r * 40 + cc * 8] =
                *(const uint4*)(A + (size_t)(tileM + r) * lda + k0 + cc * 8);
        }
        if (BMODE == 0) {
#pragma unroll
            for (int i = 0; i < 4; ++i) {           // B^T [N,K] tile 128x32
                int e = i * 128 + tid, r = e >> 2, cc = e & 3;
                *(uint4*)&Bs[r * 40 + cc * 8] =
                    *(const uint4*)(Bm + (size_t)(tileN + r) * ldb + k0 + cc * 8);
            }
        } else {
#pragma unroll
            for (int i = 0; i < 16; ++i) {          // B [K,N] tile 32x128 -> transpose
                int e = i * 128 + tid, kk = e >> 6, pp = e & 63;
                uint_t v = *(const uint_t*)(Bm + (size_t)(k0 + kk) * ldb + tileN + pp * 2);
                Bs[(pp * 2 + 0) * 40 + kk] = (ushort_t)(v & 0xFFFFu);
                Bs[(pp * 2 + 1) * 40 + kk] = (ushort_t)(v >> 16);
            }
        }
        // prefetch next K-step tiles while this one is consumed
        if (k0 + 32 < K) {
            __builtin_prefetch(A + (size_t)(tileM + (tid >> 1)) * lda + (k0 + 32) + (tid & 1) * 16, 0, 3);
            if (BMODE == 0)
                __builtin_prefetch(Bm + (size_t)(tileN + tid) * ldb + (k0 + 32), 0, 3);
            else
                __builtin_prefetch(Bm + (size_t)(k0 + 32 + (tid >> 2)) * ldb + tileN + (tid & 3) * 32, 0, 3);
        }
        __syncthreads();
        bf16x16 a0 = ldsFrag<40>(As, wm,      0, lane);
        bf16x16 a1 = ldsFrag<40>(As, wm + 16, 0, lane);
#pragma unroll
        for (int ni = 0; ni < 4; ++ni) {
            bf16x16 bb = ldsFrag<40>(Bs, wn + ni * 16, 0, lane);
            c[0][ni] = wmma_bf16(a0, bb, c[0][ni]);
            c[1][ni] = wmma_bf16(a1, bb, c[1][ni]);
        }
        __syncthreads();
    }

    const int half = lane >> 4, nl = lane & 15;
#pragma unroll
    for (int mi = 0; mi < 2; ++mi)
#pragma unroll
        for (int ni = 0; ni < 4; ++ni)
#pragma unroll
            for (int j = 0; j < 8; ++j) {
                int r   = tileM + wm + mi * 16 + j + 8 * half;
                int col = tileN + wn + ni * 16 + nl;
                float v = c[mi][ni][j];
                if (bias) v += bias[col];
                size_t off = cbase + (size_t)r * ldc + col;
                if (Cf) {
                    if (accum) v += Cf[off];
                    Cf[off] = v;
                }
                if (Cb) Cb[off] = f2bf(v);
            }
}

// ---------------------------------------------------------------------------
// Flash attention on bf16 qkv, bf16 ao out; one wave per (16-query tile, h, b).
// ---------------------------------------------------------------------------
__global__ __launch_bounds__(32) void flash_attn_kernel(
    const ushort_t* __restrict__ qkv, ushort_t* __restrict__ ao)
{
    __shared__ ushort_t Qs[16 * 104];
    __shared__ ushort_t Ks[32 * 104];
    __shared__ ushort_t Ps[16 * 40];
    __shared__ ushort_t Vt[96 * 40];

    const int b = blockIdx.z, hh = blockIdx.y, qbase = blockIdx.x * 16;
    const int lane = threadIdx.x;
    const ushort_t* base = qkv + (size_t)b * Ss * (3 * Dd);
    const int qoff = hh * DH, koff = Dd + hh * DH, voff = 2 * Dd + hh * DH;

#pragma unroll
    for (int i = 0; i < 6; ++i) {                   // Q 16x96 (12 chunks/row)
        int e = i * 32 + lane, r = e / 12, cc = e % 12;
        *(uint4*)&Qs[r * 104 + cc * 8] =
            *(const uint4*)(base + (size_t)(qbase + r) * (3 * Dd) + qoff + cc * 8);
    }
    __syncthreads();
    bf16x16 aq[3];
#pragma unroll
    for (int c = 0; c < 3; ++c) aq[c] = ldsFrag<104>(Qs, 0, c * 32, lane);

    floatx8 z = {0.f,0.f,0.f,0.f,0.f,0.f,0.f,0.f};
    floatx8 acc[6]; float mrow[8], lsum[8];
#pragma unroll
    for (int n = 0; n < 6; ++n) acc[n] = z;
#pragma unroll
    for (int j = 0; j < 8; ++j) { mrow[j] = -1e30f; lsum[j] = 0.f; }
    const float scale = 0.10206207261596575f;       // 1/sqrt(96)
    const int half = lane >> 4, nl = lane & 15;

    for (int t = 0; t < Ss / 32; ++t) {
        int kb0 = t * 32;
#pragma unroll
        for (int i = 0; i < 12; ++i) {              // K tile 32x96
            int e = i * 32 + lane, r = e / 12, cc = e % 12;
            *(uint4*)&Ks[r * 104 + cc * 8] =
                *(const uint4*)(base + (size_t)(kb0 + r) * (3 * Dd) + koff + cc * 8);
        }
#pragma unroll
        for (int i = 0; i < 48; ++i) {              // V tile 32x96 -> Vt[96][32]
            int e = i * 32 + lane, key = e / 48, pp = e % 48;
            uint_t v = *(const uint_t*)(base + (size_t)(kb0 + key) * (3 * Dd) + voff + pp * 2);
            Vt[(pp * 2 + 0) * 40 + key] = (ushort_t)(v & 0xFFFFu);
            Vt[(pp * 2 + 1) * 40 + key] = (ushort_t)(v >> 16);
        }
        if (t + 1 < Ss / 32) {                      // prefetch next K/V tile
            __builtin_prefetch(base + (size_t)(kb0 + 32 + lane) * (3 * Dd) + koff, 0, 3);
            __builtin_prefetch(base + (size_t)(kb0 + 32 + lane) * (3 * Dd) + voff, 0, 3);
        }
        __syncthreads();

        floatx8 s[2];
#pragma unroll
        for (int nt = 0; nt < 2; ++nt) {
            floatx8 cc = z;
#pragma unroll
            for (int c = 0; c < 3; ++c)
                cc = wmma_bf16(aq[c], ldsFrag<104>(Ks, nt * 16, c * 32, lane), cc);
#pragma unroll
            for (int j = 0; j < 8; ++j) cc[j] *= scale;
            s[nt] = cc;
        }
        float corr[8];
#pragma unroll
        for (int j = 0; j < 8; ++j) {
            float tm = halfMax(fmaxf(s[0][j], s[1][j]));
            float mn = fmaxf(mrow[j], tm);
            corr[j] = __expf(mrow[j] - mn);
            s[0][j] = __expf(s[0][j] - mn);
            s[1][j] = __expf(s[1][j] - mn);
            float ps = halfSum(s[0][j] + s[1][j]);
            lsum[j] = lsum[j] * corr[j] + ps;
            mrow[j] = mn;
        }
#pragma unroll
        for (int n = 0; n < 6; ++n)
#pragma unroll
            for (int j = 0; j < 8; ++j) acc[n][j] *= corr[j];
#pragma unroll
        for (int nt = 0; nt < 2; ++nt)
#pragma unroll
            for (int j = 0; j < 8; ++j)
                Ps[(j + 8 * half) * 40 + nt * 16 + nl] = f2bf(s[nt][j]);
        __syncthreads();
        bf16x16 pf = ldsFrag<40>(Ps, 0, 0, lane);
#pragma unroll
        for (int n = 0; n < 6; ++n)
            acc[n] = wmma_bf16(pf, ldsFrag<40>(Vt, n * 16, 0, lane), acc[n]);
        __syncthreads();
    }
#pragma unroll
    for (int n = 0; n < 6; ++n)
#pragma unroll
        for (int j = 0; j < 8; ++j) {
            int r = qbase + j + 8 * half;
            int col = hh * DH + n * 16 + nl;
            ao[((size_t)b * Ss + r) * Dd + col] = f2bf(acc[n][j] / lsum[j]);
        }
}

// ---------------------------------------------------------------------------
// Elementwise helpers
// ---------------------------------------------------------------------------
__global__ void copy_kernel(const float* __restrict__ src, float* __restrict__ dst, long n) {
    long i = (long)blockIdx.x * blockDim.x + threadIdx.x;
    if (i < n) dst[i] = src[i];
}
__global__ void cvt_kernel(const float* __restrict__ src, ushort_t* __restrict__ dst, long n) {
    long i = (long)blockIdx.x * blockDim.x + threadIdx.x;
    if (i < n) dst[i] = f2bf(src[i]);
}
__global__ void zero_kernel(float* __restrict__ p, long n) {
    long i = (long)blockIdx.x * blockDim.x + threadIdx.x;
    if (i < n) p[i] = 0.f;
}

__global__ __launch_bounds__(256) void topk_kernel(const float* __restrict__ sim,
                                                   int* __restrict__ outIdx) {
    __shared__ float row[Ss];
    __shared__ float rv[256];
    __shared__ int   ri[256];
    const int b = blockIdx.y, s = blockIdx.x, tid = threadIdx.x;
    const float* rp = sim + ((size_t)b * Ss + s) * Ss;
#pragma unroll
    for (int i = 0; i < Ss / 256; ++i) row[tid + i * 256] = rp[tid + i * 256];
    __syncthreads();
    for (int k = 0; k < KNN; ++k) {
        float bv = -3.0e38f; int bi = 0;
#pragma unroll
        for (int i = 0; i < Ss / 256; ++i) {
            int c = tid + i * 256; float v = row[c];
            if (v > bv) { bv = v; bi = c; }
        }
        rv[tid] = bv; ri[tid] = bi; __syncthreads();
        for (int st = 128; st > 0; st >>= 1) {
            if (tid < st) {
                if (rv[tid + st] > rv[tid] ||
                    (rv[tid + st] == rv[tid] && ri[tid + st] < ri[tid])) {
                    rv[tid] = rv[tid + st]; ri[tid] = ri[tid + st];
                }
            }
            __syncthreads();
        }
        if (tid == 0) {
            outIdx[((size_t)b * Ss + s) * KNN + k] = ri[0];
            row[ri[0]] = -3.0e38f;
        }
        __syncthreads();
    }
}

__global__ void scatter_kernel(const int* __restrict__ idx, float* __restrict__ A) {
    int i = blockIdx.x * blockDim.x + threadIdx.x;
    if (i >= Bb * Ss * KNN) return;
    int k = i % KNN, s = (i / KNN) % Ss, b = i / (KNN * Ss);
    int t = idx[i];
    float* Ab = A + (size_t)b * Ss * Ss;
    Ab[(size_t)s * Ss + t] = 1.f;
    Ab[(size_t)t * Ss + s] = 1.f;
    if (k == 0) Ab[(size_t)s * Ss + s] = 1.f;
}

__global__ __launch_bounds__(256) void rowsum_kernel(const float* __restrict__ A,
                                                     float* __restrict__ dinv) {
    __shared__ float red[256];
    const int b = blockIdx.y, s = blockIdx.x, tid = threadIdx.x;
    const float* rp = A + ((size_t)b * Ss + s) * Ss;
    float sum = 0.f;
#pragma unroll
    for (int i = 0; i < Ss / 256; ++i) sum += rp[tid + i * 256];
    red[tid] = sum; __syncthreads();
    for (int st = 128; st > 0; st >>= 1) {
        if (tid < st) red[tid] += red[tid + st];
        __syncthreads();
    }
    if (tid == 0) dinv[b * Ss + s] = rsqrtf(red[0]);
}

__global__ void normadj_kernel(float* __restrict__ A, const float* __restrict__ dinv) {
    long i = (long)blockIdx.x * blockDim.x + threadIdx.x;
    if (i >= (long)Bb * Ss * Ss) return;
    int t = (int)(i & (Ss - 1));
    int s = (int)((i >> 10) & (Ss - 1));
    int b = (int)(i >> 20);
    A[i] *= dinv[b * Ss + s] * dinv[b * Ss + t];
}

// ---------------------------------------------------------------------------
// gate = sigmoid(gatelin); h = LN(gate*gcn + (1-gate)*ao2 + h); also hb = bf16(h)
// ---------------------------------------------------------------------------
__global__ __launch_bounds__(256) void fuse_ln_kernel(
    const float* __restrict__ gatelin, const float* __restrict__ gcn,
    const float* __restrict__ ao2, float* __restrict__ h, ushort_t* __restrict__ hb,
    const float* __restrict__ g, const float* __restrict__ beta)
{
    __shared__ float red[256];
    const int row = blockIdx.x, tid = threadIdx.x;
    const size_t base = (size_t)row * Dd;
    float f[Dd / 256];
    float sum = 0.f;
#pragma unroll
    for (int i = 0; i < Dd / 256; ++i) {
        int c = tid + i * 256;
        float gl = gatelin[base + c];
        float gg = 1.f / (1.f + __expf(-gl));
        float v = gg * gcn[base + c] + (1.f - gg) * ao2[base + c] + h[base + c];
        f[i] = v; sum += v;
    }
    red[tid] = sum; __syncthreads();
    for (int st = 128; st > 0; st >>= 1) { if (tid < st) red[tid] += red[tid + st]; __syncthreads(); }
    float mu = red[0] / Dd; __syncthreads();
    float vs = 0.f;
#pragma unroll
    for (int i = 0; i < Dd / 256; ++i) { float d = f[i] - mu; vs += d * d; }
    red[tid] = vs; __syncthreads();
    for (int st = 128; st > 0; st >>= 1) { if (tid < st) red[tid] += red[tid + st]; __syncthreads(); }
    float inv = rsqrtf(red[0] / Dd + 1e-5f); __syncthreads();
#pragma unroll
    for (int i = 0; i < Dd / 256; ++i) {
        int c = tid + i * 256;
        float v = (f[i] - mu) * inv * g[c] + beta[c];
        h[base + c]  = v;
        hb[base + c] = f2bf(v);
    }
}

// ---------------------------------------------------------------------------
extern "C" void kernel_launch(void* const* d_in, const int* in_sizes, int n_in,
                              void* d_out, int out_size, void* d_ws, size_t ws_size,
                              hipStream_t stream) {
    const float* x          = (const float*)d_in[0];
    const float* gcn_w      = (const float*)d_in[1];
    const float* gcn_b      = (const float*)d_in[2];
    const float* attn_in_w  = (const float*)d_in[3];
    const float* attn_in_b  = (const float*)d_in[4];
    const float* attn_out_w = (const float*)d_in[5];
    const float* attn_out_b = (const float*)d_in[6];
    const float* ln_g       = (const float*)d_in[7];
    const float* ln_b       = (const float*)d_in[8];
    const float* gate_w     = (const float*)d_in[9];
    const float* gate_b     = (const float*)d_in[10];
    const float* proj_w     = (const float*)d_in[11];
    const float* proj_b     = (const float*)d_in[12];

    const long BSS  = (long)Bb * Ss * Ss;      // 8388608
    const long BSD  = (long)Bb * Ss * Dd;      // 6291456
    const long BS3D = (long)Bb * Ss * 3 * Dd;  // 18874368
    const long nWgcn  = (long)LL * Dd * Dd;
    const long nWin   = (long)LL * 3 * Dd * Dd;
    const long nWout  = (long)LL * Dd * Dd;
    const long nWgate = (long)LL * Dd * 2 * Dd;
    const long nWproj = (long)Dd * Dd;

    // f32 region
    float* An      = (float*)d_ws;
    float* h       = An + BSS;
    float* gcn     = h + BSD;
    float* ao2     = gcn + BSD;
    float* gatelin = ao2 + BSD;
    // bf16 region
    ushort_t* hb    = (ushort_t*)(gatelin + BSD);
    ushort_t* Anb   = hb + BSD;
    ushort_t* tmpb  = Anb + BSS;          // Gb, then ao (bf16)
    ushort_t* gcnb  = tmpb + BSD;
    ushort_t* ao2b  = gcnb + BSD;
    ushort_t* qkvb  = ao2b + BSD;
    ushort_t* wgcnb = qkvb + BS3D;
    ushort_t* winb  = wgcnb + nWgcn;
    ushort_t* woutb = winb + nWin;
    ushort_t* wgateb= woutb + nWout;
    ushort_t* wprojb= wgateb + nWgate;
    int*   idx  = (int*)(wprojb + nWproj);
    float* dinv = (float*)(idx + Bb * Ss * KNN);

    // h = x (f32 + bf16)
    copy_kernel<<<dim3((BSD + 255) / 256), 256, 0, stream>>>(x, h, BSD);
    cvt_kernel<<<dim3((BSD + 255) / 256), 256, 0, stream>>>(x, hb, BSD);
    // weights -> bf16 (once)
    cvt_kernel<<<dim3((nWgcn  + 255) / 256), 256, 0, stream>>>(gcn_w,      wgcnb,  nWgcn);
    cvt_kernel<<<dim3((nWin   + 255) / 256), 256, 0, stream>>>(attn_in_w,  winb,   nWin);
    cvt_kernel<<<dim3((nWout  + 255) / 256), 256, 0, stream>>>(attn_out_w, woutb,  nWout);
    cvt_kernel<<<dim3((nWgate + 255) / 256), 256, 0, stream>>>(gate_w,     wgateb, nWgate);
    cvt_kernel<<<dim3((nWproj + 255) / 256), 256, 0, stream>>>(proj_w,     wprojb, nWproj);

    // sim = x @ x^T  -> An (f32)
    gemm_bf16_kernel<0><<<dim3(Ss / 128, Ss / 64, Bb), 128, 0, stream>>>(
        hb, hb, An, nullptr, nullptr, Dd, Dd, Ss, Dd,
        (long)Ss * Dd, (long)Ss * Dd, (long)Ss * Ss, 0);

    topk_kernel<<<dim3(Ss, Bb), 256, 0, stream>>>(An, idx);
    zero_kernel<<<dim3((BSS + 255) / 256), 256, 0, stream>>>(An, BSS);
    scatter_kernel<<<dim3((Bb * Ss * KNN + 255) / 256), 256, 0, stream>>>(idx, An);
    rowsum_kernel<<<dim3(Ss, Bb), 256, 0, stream>>>(An, dinv);
    normadj_kernel<<<dim3((BSS + 255) / 256), 256, 0, stream>>>(An, dinv);
    cvt_kernel<<<dim3((BSS + 255) / 256), 256, 0, stream>>>(An, Anb, BSS);

    const int M = Bb * Ss;   // 8192
    for (int l = 0; l < LL; ++l) {
        // G(bf16) = h @ gcn_w[l]^T
        gemm_bf16_kernel<0><<<dim3(Dd / 128, M / 64, 1), 128, 0, stream>>>(
            hb, wgcnb + (size_t)l * Dd * Dd, nullptr, tmpb, nullptr,
            Dd, Dd, Dd, Dd, 0, 0, 0, 0);
        // gcn = An @ G + gcn_b[l]  (per-batch NN) -> f32 + bf16
        gemm_bf16_kernel<1><<<dim3(Dd / 128, Ss / 64, Bb), 128, 0, stream>>>(
            Anb, tmpb, gcn, gcnb, gcn_b + (size_t)l * Dd,
            Ss, Dd, Dd, Ss, (long)Ss * Ss, (long)Ss * Dd, (long)Ss * Dd, 0);
        // qkv(bf16) = h @ attn_in_w[l]^T + attn_in_b[l]
        gemm_bf16_kernel<0><<<dim3(3 * Dd / 128, M / 64, 1), 128, 0, stream>>>(
            hb, winb + (size_t)l * 3 * Dd * Dd, nullptr, qkvb,
            attn_in_b + (size_t)l * 3 * Dd,
            Dd, Dd, 3 * Dd, Dd, 0, 0, 0, 0);
        // ao(bf16) = attention(qkv) -> tmpb (G is dead)
        flash_attn_kernel<<<dim3(Ss / 16, Hh, Bb), 32, 0, stream>>>(qkvb, tmpb);
        // ao2 = ao @ attn_out_w[l]^T + attn_out_b[l] -> f32 + bf16
        gemm_bf16_kernel<0><<<dim3(Dd / 128, M / 64, 1), 128, 0, stream>>>(
            tmpb, woutb + (size_t)l * Dd * Dd, ao2, ao2b,
            attn_out_b + (size_t)l * Dd,
            Dd, Dd, Dd, Dd, 0, 0, 0, 0);
        // gatelin = gcn @ gate_w[l][:, :D]^T + gate_b[l]
        gemm_bf16_kernel<0><<<dim3(Dd / 128, M / 64, 1), 128, 0, stream>>>(
            gcnb, wgateb + (size_t)l * Dd * 2 * Dd, gatelin, nullptr,
            gate_b + (size_t)l * Dd,
            Dd, 2 * Dd, Dd, Dd, 0, 0, 0, 0);
        // gatelin += ao2 @ gate_w[l][:, D:]^T
        gemm_bf16_kernel<0><<<dim3(Dd / 128, M / 64, 1), 128, 0, stream>>>(
            ao2b, wgateb + (size_t)l * Dd * 2 * Dd + Dd, gatelin, nullptr, nullptr,
            Dd, 2 * Dd, Dd, Dd, 0, 0, 0, 1);
        // h = LN(sig(gatelin)*gcn + (1-sig)*ao2 + h), dual write f32+bf16
        fuse_ln_kernel<<<dim3(M), 256, 0, stream>>>(
            gatelin, gcn, ao2, h, hb,
            ln_g + (size_t)l * Dd, ln_b + (size_t)l * Dd);
    }

    // out = h @ proj_w^T + proj_b
    gemm_bf16_kernel<0><<<dim3(Dd / 128, M / 64, 1), 128, 0, stream>>>(
        hb, wprojb, (float*)d_out, nullptr, proj_b,
        Dd, Dd, Dd, Dd, 0, 0, 0, 0);
}
